// MultiHeadAttention_20117626814570
// MI455X (gfx1250) — matrix operand order repaired and checked by reading, not compiled, not run on hardware
//
#include <hip/hip_runtime.h>
#include <hip/hip_bf16.h>

// MHA forward for B=4, S=2048, H=16, DK=64, D=1024 on gfx1250 (MI455X).
// f32 -> bf16 once, all matmuls on v_wmma_f32_16x16x32_bf16 (f32 accum).
// GEMMs: double-buffered LDS, single barrier/iter, globals issued before the
// WMMA burst. bf16 tiles (attention K/V, out-proj A) staged with
// global_load_async_to_lds_b128 (ASYNCcnt). Softmax row reductions done on
// the VALU with DPP row_ror rotate-butterflies (no ds_bpermute).

#define Bn 4
#define Sn 2048
#define Hn 16
#define DKn 64
#define Dn 1024
#define Mn (Bn * Sn) // 8192

typedef __attribute__((ext_vector_type(16))) __bf16 bf16x16;
typedef __attribute__((ext_vector_type(8))) float floatx8;

__device__ __forceinline__ floatx8 wmma_bf16(bf16x16 a, bf16x16 b, floatx8 c) {
  return __builtin_amdgcn_wmma_f32_16x16x32_bf16(false, a, false, b, (short)0, c,
                                                 false, false);
}

// Async 16B global -> LDS copy (per lane). LDS address = low 32 bits of the
// generic pointer (flat LDS aperture addresses are wave-relative offsets).
__device__ __forceinline__ void async_copy_b128(const void* gptr, void* lptr) {
  unsigned loff = (unsigned)(unsigned long long)lptr;
  unsigned long long ga = (unsigned long long)gptr;
  asm volatile("global_load_async_to_lds_b128 %0, %1, off"
               :: "v"(loff), "v"(ga) : "memory");
}
__device__ __forceinline__ void wait_async0() {
  asm volatile("s_wait_asynccnt 0x0" ::: "memory");
}

// ---- DPP16 row (16-lane) reductions: rotate-butterfly, VALU only ----------
template <int N>
__device__ __forceinline__ float dpp_ror(float x) {
  return __int_as_float(__builtin_amdgcn_update_dpp(
      0, __float_as_int(x), 0x120 | N, 0xf, 0xf, true));  // row_ror:N
}
__device__ __forceinline__ float row_max16(float x) {
  x = fmaxf(x, dpp_ror<1>(x));
  x = fmaxf(x, dpp_ror<2>(x));
  x = fmaxf(x, dpp_ror<4>(x));
  x = fmaxf(x, dpp_ror<8>(x));
  return x;
}
__device__ __forceinline__ float row_sum16(float x) {
  x += dpp_ror<1>(x);
  x += dpp_ror<2>(x);
  x += dpp_ror<4>(x);
  x += dpp_ror<8>(x);
  return x;
}

// ---- Fragment loaders (ISA 7.12.2 layouts, wave32) -------------------------
// A 16x32 bf16: lane l -> m=l&15, kh=l>>4; VGPR i: K = 8*kh + (i<4?2i:16+2(i-4)).
template <int STRIDE>
__device__ __forceinline__ bf16x16 load_frag_a(const __bf16* base, int row0, int k0) {
  int l = threadIdx.x & 31;
  int m = l & 15, kh = l >> 4;
  union { bf16x16 v; unsigned int u[8]; } f;
  const __bf16* p = base + (row0 + m) * STRIDE + k0 + 8 * kh;
  f.u[0] = *(const unsigned int*)(p + 0);
  f.u[1] = *(const unsigned int*)(p + 2);
  f.u[2] = *(const unsigned int*)(p + 4);
  f.u[3] = *(const unsigned int*)(p + 6);
  f.u[4] = *(const unsigned int*)(p + 16);
  f.u[5] = *(const unsigned int*)(p + 18);
  f.u[6] = *(const unsigned int*)(p + 20);
  f.u[7] = *(const unsigned int*)(p + 22);
  return f.v;
}

// B 32x16 bf16 stored as base[n][k]: lane l -> n=l&15, kh=l>>4;
// VGPR i: K = 8*kh + (i<4?2i:16+2(i-4)) -> 2x 16 contiguous bytes (2x ds_load_b128).
template <int STRIDE>
__device__ __forceinline__ bf16x16 load_frag_b(const __bf16* base, int row0, int k0) {
  int l = threadIdx.x & 31;
  int n = l & 15, kh = l >> 4;
  union { bf16x16 v; uint4 q[2]; } f;
  const __bf16* p = base + (row0 + n) * STRIDE + k0 + 8 * kh;
  f.q[0] = *(const uint4*)(p);
  f.q[1] = *(const uint4*)(p + 16);
  return f.v;
}

// 128x128 block tile, 8 waves, wave = 64x32 -> 8 WMMA accumulators.
__device__ __forceinline__ void mma_tile(const __bf16* A, const __bf16* Bt,
                                         int wr, int wc, floatx8 (&acc)[4][2]) {
  bf16x16 af[4], bfr[2];
#pragma unroll
  for (int i = 0; i < 4; ++i) af[i] = load_frag_a<40>(A, wr * 64 + 16 * i, 0);
#pragma unroll
  for (int j = 0; j < 2; ++j) bfr[j] = load_frag_b<40>(Bt, wc * 32 + 16 * j, 0);
#pragma unroll
  for (int i = 0; i < 4; ++i)
#pragma unroll
    for (int j = 0; j < 2; ++j) acc[i][j] = wmma_bf16(af[i], bfr[j], acc[i][j]);
}

// ---- QKV projection: X[8192,1024]f32 @ W + b -> bf16 -----------------------
// WHICH: 0 = Q (scaled 1/8, [b,h,s,d]); 1 = K ([b,h,s,d]); 2 = V ([b,h,d,s]).
template <int WHICH>
__global__ void __launch_bounds__(256) qkv_proj_kernel(
    const float* __restrict__ X, const float* __restrict__ W,
    const float* __restrict__ bias, __bf16* __restrict__ out) {
  __shared__ __align__(16) __bf16 lds_A[2][128 * 40];  // [m][k]
  __shared__ __align__(16) __bf16 lds_B[2][128 * 40];  // [n][k] (W transposed)

  const int m0 = blockIdx.y * 128;
  const int n0 = blockIdx.x * 128;
  const int tid = threadIdx.x;
  const int wid = tid >> 5;
  const int wr = wid >> 2, wc = wid & 3;

  const int arow = tid >> 1, ach = (tid & 1) * 16;   // A: 2 thr/row, 16 cols
  const int bkk = tid >> 3, bnq = (tid & 7) * 16;    // B: 8 thr/k-row, 16 cols

  float4 ra[4], rb[4];
  floatx8 acc[4][2] = {};

  auto gload = [&](int kt) {
    const float* ga = X + (size_t)(m0 + arow) * Dn + kt + ach;
    const float* gb = W + (size_t)(kt + bkk) * Dn + n0 + bnq;
#pragma unroll
    for (int j = 0; j < 4; ++j) {
      ra[j] = *(const float4*)(ga + 4 * j);
      rb[j] = *(const float4*)(gb + 4 * j);
    }
  };
  auto lstore = [&](int buf) {
    __bf16* sa = &lds_A[buf][arow * 40 + ach];
#pragma unroll
    for (int j = 0; j < 4; ++j) {
      union { __bf16 h[2]; unsigned u; } p0, p1;
      p0.h[0] = (__bf16)ra[j].x; p0.h[1] = (__bf16)ra[j].y;
      p1.h[0] = (__bf16)ra[j].z; p1.h[1] = (__bf16)ra[j].w;
      *(unsigned*)(sa + 4 * j) = p0.u;
      *(unsigned*)(sa + 4 * j + 2) = p1.u;
      lds_B[buf][(bnq + 4 * j + 0) * 40 + bkk] = (__bf16)rb[j].x;
      lds_B[buf][(bnq + 4 * j + 1) * 40 + bkk] = (__bf16)rb[j].y;
      lds_B[buf][(bnq + 4 * j + 2) * 40 + bkk] = (__bf16)rb[j].z;
      lds_B[buf][(bnq + 4 * j + 3) * 40 + bkk] = (__bf16)rb[j].w;
    }
  };

  gload(0);
  lstore(0);
#pragma unroll 1
  for (int it = 0; it < Dn / 32; ++it) {
    const int buf = it & 1;
    __syncthreads();
    if (it + 1 < Dn / 32) gload((it + 1) * 32);     // in flight during WMMAs
    mma_tile(lds_A[buf], lds_B[buf], wr, wc, acc);
    if (it + 1 < Dn / 32) lstore(buf ^ 1);
  }

  const int l = tid & 31;
  const int nlo = l & 15, hi = l >> 4;
#pragma unroll
  for (int i = 0; i < 4; ++i) {
#pragma unroll
    for (int j = 0; j < 2; ++j) {
      int col = n0 + wc * 32 + j * 16 + nlo;
      float bval = bias[col];
      int h = col >> 6, d = col & 63;
#pragma unroll
      for (int r = 0; r < 8; ++r) {
        int grow = m0 + wr * 64 + i * 16 + r + 8 * hi;
        int b = grow >> 11, s = grow & (Sn - 1);
        float v = acc[i][j][r] + bval;
        if constexpr (WHICH == 0)
          out[((size_t)(b * Hn + h) * Sn + s) * DKn + d] = (__bf16)(v * 0.125f);
        else if constexpr (WHICH == 1)
          out[((size_t)(b * Hn + h) * Sn + s) * DKn + d] = (__bf16)v;
        else
          out[((size_t)(b * Hn + h) * DKn + d) * Sn + s] = (__bf16)v;
      }
    }
  }
}

// ---- Flash attention: per (b,h), 64 q rows/block, k-tiles of 32 ------------
__global__ void __launch_bounds__(128) attn_kernel(
    const __bf16* __restrict__ Qh, const __bf16* __restrict__ Kh,
    const __bf16* __restrict__ Vt, __bf16* __restrict__ AO) {
  __shared__ __align__(16) __bf16 lds_K[2][32 * 80];      // [s_local][d]
  __shared__ __align__(16) __bf16 lds_V[2][64 * 40];      // [d][s_local]
  __shared__ __align__(16) __bf16 lds_P[4 * 16 * 40];     // per-wave P 16x32

  const int bh = blockIdx.y;
  const int q0 = blockIdx.x * 64;
  const int tid = threadIdx.x;
  const int wid = tid >> 5;
  const int l = tid & 31;
  const int nlo = l & 15, hi = l >> 4;

  const __bf16* qbase = Qh + ((size_t)bh * Sn + q0 + wid * 16) * DKn;
  bf16x16 qf0 = load_frag_a<DKn>(qbase, 0, 0);
  bf16x16 qf1 = load_frag_a<DKn>(qbase, 0, 32);

  float mrow[8], lrow[8];
  floatx8 oacc[4] = {};
#pragma unroll
  for (int r = 0; r < 8; ++r) { mrow[r] = -3.0e38f; lrow[r] = 0.0f; }

  __bf16* pb = lds_P + wid * 16 * 40;

  // async stage of one 32-wide K/V tile pair (4x b128 per thread)
  auto issue_tiles = [&](int s0, int buf) {
    {
      int row = tid >> 2, dq = (tid & 3) * 16;  // K: 32 x 64
      const __bf16* g = Kh + ((size_t)bh * Sn + s0 + row) * DKn + dq;
      __bf16* s = &lds_K[buf][row * 80 + dq];
      async_copy_b128(g, s);
      async_copy_b128(g + 8, s + 8);
    }
    {
      int dr = tid >> 1, sq = (tid & 1) * 16;   // V: 64 x 32 (transposed src)
      const __bf16* g = Vt + ((size_t)bh * DKn + dr) * Sn + s0 + sq;
      __bf16* s = &lds_V[buf][dr * 40 + sq];
      async_copy_b128(g, s);
      async_copy_b128(g + 8, s + 8);
    }
  };

  issue_tiles(0, 0);
#pragma unroll 1
  for (int it = 0; it < Sn / 32; ++it) {
    const int buf = it & 1;
    wait_async0();                 // my tile writes landed (compiler can't see asm)
    __syncthreads();               // everyone's landed; prev reads of buf^1 done
    if (it + 1 < Sn / 32) issue_tiles((it + 1) * 32, buf ^ 1);

    // logits 16(q) x 32(s) = Q(16x64) @ K^T
    floatx8 sacc[2] = {};
#pragma unroll
    for (int t = 0; t < 2; ++t) {
      bf16x16 qf = t ? qf1 : qf0;
#pragma unroll
      for (int j = 0; j < 2; ++j) {
        bf16x16 kf = load_frag_b<80>(lds_K[buf], j * 16, t * 32);
        sacc[j] = wmma_bf16(qf, kf, sacc[j]);
      }
    }

    // online softmax; DPP16 row reductions (VALU only, no LDS traffic)
    float alpha[8];
#pragma unroll
    for (int r = 0; r < 8; ++r) {
      float cmax = row_max16(fmaxf(sacc[0][r], sacc[1][r]));
      float mnew = fmaxf(mrow[r], cmax);
      alpha[r] = __expf(mrow[r] - mnew);
      mrow[r] = mnew;
      float p0 = __expf(sacc[0][r] - mnew);
      float p1 = __expf(sacc[1][r] - mnew);
      int row = r + 8 * hi;
      pb[row * 40 + nlo] = (__bf16)p0;        // C-layout -> A-layout via LDS
      pb[row * 40 + 16 + nlo] = (__bf16)p1;
      lrow[r] = lrow[r] * alpha[r] + row_sum16(p0 + p1);
    }
#pragma unroll
    for (int jj = 0; jj < 4; ++jj)
#pragma unroll
      for (int r = 0; r < 8; ++r) oacc[jj][r] = oacc[jj][r] * alpha[r];

    asm volatile("s_wait_dscnt 0x0" ::: "memory");  // same-wave P RAW

    bf16x16 pf = load_frag_a<40>(pb, 0, 0);
#pragma unroll
    for (int jj = 0; jj < 4; ++jj) {
      bf16x16 vf = load_frag_b<40>(lds_V[buf], jj * 16, 0);
      oacc[jj] = wmma_bf16(pf, vf, oacc[jj]);
    }
  }

  const int b = bh >> 4, h = bh & 15;
#pragma unroll
  for (int jj = 0; jj < 4; ++jj) {
    int d = jj * 16 + nlo;
#pragma unroll
    for (int r = 0; r < 8; ++r) {
      int qrow = q0 + wid * 16 + r + 8 * hi;
      AO[((size_t)b * Sn + qrow) * Dn + h * 64 + d] =
          (__bf16)(oacc[jj][r] / lrow[r]);
    }
  }
}

// ---- Output projection: AO[8192,1024]bf16 @ Wo + bo -> f32 -----------------
__global__ void __launch_bounds__(256) out_proj_kernel(
    const __bf16* __restrict__ AO, const float* __restrict__ Wo,
    const float* __restrict__ bo, float* __restrict__ out) {
  __shared__ __align__(16) __bf16 lds_A[2][128 * 40];
  __shared__ __align__(16) __bf16 lds_B[2][128 * 40];

  const int m0 = blockIdx.y * 128;
  const int n0 = blockIdx.x * 128;
  const int tid = threadIdx.x;
  const int wid = tid >> 5;
  const int wr = wid >> 2, wc = wid & 3;

  const int arow = tid >> 1, ach = (tid & 1) * 16;
  const int bkk = tid >> 3, bnq = (tid & 7) * 16;

  float4 rb[4];
  floatx8 acc[4][2] = {};

  auto issue_A = [&](int kt, int buf) {  // bf16 A tile: async, no VGPR trip
    const __bf16* g = AO + (size_t)(m0 + arow) * Dn + kt + ach;
    __bf16* s = &lds_A[buf][arow * 40 + ach];
    async_copy_b128(g, s);
    async_copy_b128(g + 8, s + 8);
  };
  auto gload_B = [&](int kt) {
    const float* gb = Wo + (size_t)(kt + bkk) * Dn + n0 + bnq;
#pragma unroll
    for (int j = 0; j < 4; ++j) rb[j] = *(const float4*)(gb + 4 * j);
  };
  auto store_B = [&](int buf) {
#pragma unroll
    for (int j = 0; j < 4; ++j) {
      lds_B[buf][(bnq + 4 * j + 0) * 40 + bkk] = (__bf16)rb[j].x;
      lds_B[buf][(bnq + 4 * j + 1) * 40 + bkk] = (__bf16)rb[j].y;
      lds_B[buf][(bnq + 4 * j + 2) * 40 + bkk] = (__bf16)rb[j].z;
      lds_B[buf][(bnq + 4 * j + 3) * 40 + bkk] = (__bf16)rb[j].w;
    }
  };

  issue_A(0, 0);
  gload_B(0);
  store_B(0);
#pragma unroll 1
  for (int it = 0; it < Dn / 32; ++it) {
    const int buf = it & 1;
    wait_async0();
    __syncthreads();
    if (it + 1 < Dn / 32) {
      issue_A((it + 1) * 32, buf ^ 1);
      gload_B((it + 1) * 32);
    }
    mma_tile(lds_A[buf], lds_B[buf], wr, wc, acc);
    if (it + 1 < Dn / 32) store_B(buf ^ 1);
  }

  const int l = tid & 31;
  const int nlo = l & 15, hi = l >> 4;
#pragma unroll
  for (int i = 0; i < 4; ++i) {
#pragma unroll
    for (int j = 0; j < 2; ++j) {
      int col = n0 + wc * 32 + j * 16 + nlo;
      float bval = bo[col];
#pragma unroll
      for (int r = 0; r < 8; ++r) {
        int grow = m0 + wr * 64 + i * 16 + r + 8 * hi;
        out[(size_t)grow * Dn + col] = acc[i][j][r] + bval;
      }
    }
  }
}

extern "C" void kernel_launch(void* const* d_in, const int* in_sizes, int n_in,
                              void* d_out, int out_size, void* d_ws, size_t ws_size,
                              hipStream_t stream) {
  // setup_inputs order: query, value, key, Wq, bq, Wk, bk, Wv, bv, Wo, bo
  const float* query = (const float*)d_in[0];
  const float* value = (const float*)d_in[1];
  const float* key   = (const float*)d_in[2];
  const float* Wq = (const float*)d_in[3];
  const float* bq = (const float*)d_in[4];
  const float* Wk = (const float*)d_in[5];
  const float* bk = (const float*)d_in[6];
  const float* Wv = (const float*)d_in[7];
  const float* bv = (const float*)d_in[8];
  const float* Wo = (const float*)d_in[9];
  const float* bo = (const float*)d_in[10];

  const size_t elems = (size_t)Mn * Dn;
  __bf16* Qh = (__bf16*)d_ws;   // [B,H,S,DK]
  __bf16* Kh = Qh + elems;      // [B,H,S,DK]
  __bf16* Vt = Kh + elems;      // [B,H,DK,S]
  __bf16* AO = Vt + elems;      // [B,S,D]

  dim3 gg(Dn / 128, Mn / 128);
  qkv_proj_kernel<0><<<gg, dim3(256), 0, stream>>>(query, Wq, bq, Qh);
  qkv_proj_kernel<1><<<gg, dim3(256), 0, stream>>>(key, Wk, bk, Kh);
  qkv_proj_kernel<2><<<gg, dim3(256), 0, stream>>>(value, Wv, bv, Vt);
  attn_kernel<<<dim3(Sn / 64, Bn * Hn), dim3(128), 0, stream>>>(Qh, Kh, Vt, AO);
  out_proj_kernel<<<gg, dim3(256), 0, stream>>>(AO, Wo, bo, (float*)d_out);
}
